// MixtureOfExperts_17643725652340
// MI455X (gfx1250) — matrix operand order repaired and checked by reading, not compiled, not run on hardware
//
#include <hip/hip_runtime.h>
#include <cstdint>

// Problem constants (from reference): T,H,F,E,K
#define T_   1024
#define H_   1024
#define F_   2048
#define E_   8
#define KSEL 2
#define CAP  (T_ * KSEL)   // worst-case rows per expert

// Tiling
#define BM 64              // pair-rows per block
#define BK 32              // K depth per WMMA step (bf16 16x16x32)
#define PA 40              // LDS pitch (ushorts) for A tile (padded vs 32)
#define PB 40              // LDS pitch (ushorts) for B tile

typedef __bf16 bf16x16 __attribute__((ext_vector_type(16)));
typedef float  floatx8 __attribute__((ext_vector_type(8)));
typedef int    v4i_async __attribute__((vector_size(16)));

struct alignas(16) U4 { unsigned int x, y, z, w; };
struct alignas(16) F4 { float x, y, z, w; };
union Frag { U4 q[2]; bf16x16 v; };

__device__ __forceinline__ unsigned short f2bf(float f) {
  // round-to-nearest-even f32 -> bf16
  unsigned int u = __builtin_bit_cast(unsigned int, f);
  u += 0x7FFFu + ((u >> 16) & 1u);
  return (unsigned short)(u >> 16);
}

// 16B global -> LDS async copy (ASYNCcnt-tracked, no VGPR data movement)
__device__ __forceinline__ void async_cp16(void* lds_dst, const void* glb_src) {
  __builtin_amdgcn_global_load_async_to_lds_b128(
      (v4i_async*)glb_src,
      (__attribute__((address_space(3))) v4i_async*)lds_dst, 0, 0);
}
__device__ __forceinline__ void wait_async0() {
  asm volatile("s_wait_asynccnt 0" ::: "memory");
}

// ---------------------------------------------------------------------------
// K0: zero output, zero expert counters, convert hidden_states f32 -> bf16
// ---------------------------------------------------------------------------
__global__ __launch_bounds__(256) void moe_init(const float* __restrict__ x,
                                                float* __restrict__ out,
                                                unsigned short* __restrict__ X16,
                                                int* __restrict__ cnt) {
  int i = blockIdx.x * blockDim.x + threadIdx.x;
  int stride = gridDim.x * blockDim.x;
  for (int idx = i; idx < T_ * H_; idx += stride) {
    out[idx] = 0.0f;
    X16[idx] = f2bf(x[idx]);
  }
  if (i < E_) cnt[i] = 0;
}

// ---------------------------------------------------------------------------
// K1: route (token,k) pairs into per-expert compacted lists
// ---------------------------------------------------------------------------
__global__ __launch_bounds__(256) void moe_route(const int* __restrict__ sel,
                                                 int* __restrict__ cnt,
                                                 int* __restrict__ list) {
  int p = blockIdx.x * blockDim.x + threadIdx.x;
  if (p >= T_ * KSEL) return;
  int e = sel[p];
  int pos = atomicAdd(&cnt[e], 1);
  list[e * CAP + pos] = p;
}

// ---------------------------------------------------------------------------
// K2: GEMM1 + SiLU.  act[p, c] = silu(x@w1g^T + b1g) * (x@w1l^T + b1l)
// 64 pair-rows x 64 cols (gate AND matching lin), K over H, double-buffered.
// ---------------------------------------------------------------------------
__global__ __launch_bounds__(256) void moe_gemm1(
    const unsigned short* __restrict__ X16,   // [T][H] bf16
    const float* __restrict__ w1,             // [E][2F][H] f32
    const float* __restrict__ b1,             // [E][2F] f32
    const int* __restrict__ cnt,              // [E]
    const int* __restrict__ list,             // [E][CAP]
    unsigned short* __restrict__ act)         // [T*K][F] bf16
{
  const int e   = blockIdx.z;
  const int n_e = cnt[e];
  const int rT  = blockIdx.y;
  if (rT * BM >= n_e) return;
  const int c0  = blockIdx.x * 64;

  __shared__ unsigned short sA[2][BM * PA];    // A: 64 rows x 32 k (bf16)
  __shared__ unsigned short sB[2][128 * PB];   // B: 64 gate + 64 lin rows
  __shared__ int sPair[BM];

  const int tid = threadIdx.x;
  if (tid < BM) {
    int r = rT * BM + tid;
    sPair[tid] = (r < n_e) ? list[e * CAP + r] : -1;
  }
  __syncthreads();

  // Per-thread staging assignments, hoisted out of the k-loop.
  const int aRow = tid >> 2, aSeg = tid & 3;        // A: 16B per thread
  int pA = sPair[aRow];
  const unsigned short* aSrc =
      X16 + (size_t)((pA < 0 ? 0 : pA) >> 1) * H_ + aSeg * 8;  // token row (alias row 0 if invalid)

  const int bRow = tid >> 1, bHalf = tid & 1;       // B: 16 f32 per thread
  const int wrow = (bRow < 64) ? (c0 + bRow) : (F_ + c0 + (bRow - 64));
  const float* bSrc =
      w1 + (size_t)e * (2 * F_) * H_ + (size_t)wrow * H_ + bHalf * 16;

  const int wave = tid >> 5, lane = tid & 31;
  const int wm = wave >> 1, wn = wave & 1;
  const int half = lane >> 4, ln = lane & 15;

  floatx8 accG[2] = {};
  floatx8 accL[2] = {};

  auto stage = [&](int buf, int k0) {
    async_cp16(&sA[buf][aRow * PA + aSeg * 8], aSrc + k0);
    __builtin_prefetch(bSrc + k0 + 2 * BK, 0, 0);   // global_prefetch next-next tile
    const F4* s4 = (const F4*)(bSrc + k0);
    alignas(16) unsigned short tmp[16];
    #pragma unroll
    for (int q = 0; q < 4; ++q) {
      F4 f = s4[q];
      tmp[4 * q + 0] = f2bf(f.x); tmp[4 * q + 1] = f2bf(f.y);
      tmp[4 * q + 2] = f2bf(f.z); tmp[4 * q + 3] = f2bf(f.w);
    }
    *(U4*)(&sB[buf][bRow * PB + bHalf * 16 + 0]) = *(const U4*)(&tmp[0]);
    *(U4*)(&sB[buf][bRow * PB + bHalf * 16 + 8]) = *(const U4*)(&tmp[8]);
  };

  stage(0, 0);
  wait_async0();
  __syncthreads();

  const int nk = H_ / BK;
  for (int i = 0; i < nk; ++i) {
    const int buf = i & 1;
    if (i + 1 < nk) stage(buf ^ 1, (i + 1) * BK);

    Frag fa;
    {
      int m = wm * 16 + ln;
      fa.q[0] = *(const U4*)(&sA[buf][m * PA + half * 16]);
      fa.q[1] = *(const U4*)(&sA[buf][m * PA + half * 16 + 8]);
    }
    #pragma unroll
    for (int cf = 0; cf < 2; ++cf) {
      int ng = wn * 32 + cf * 16 + ln;     // gate row in LDS tile
      Frag fbg, fbl;
      fbg.q[0] = *(const U4*)(&sB[buf][ng * PB + half * 16 + 0]);
      fbg.q[1] = *(const U4*)(&sB[buf][ng * PB + half * 16 + 8]);
      int nl = 64 + ng;                    // matching lin row
      fbl.q[0] = *(const U4*)(&sB[buf][nl * PB + half * 16 + 0]);
      fbl.q[1] = *(const U4*)(&sB[buf][nl * PB + half * 16 + 8]);
      accG[cf] = __builtin_amdgcn_wmma_f32_16x16x32_bf16(
          false, fa.v, false, fbg.v, (short)0, accG[cf], false, false);
      accL[cf] = __builtin_amdgcn_wmma_f32_16x16x32_bf16(
          false, fa.v, false, fbl.v, (short)0, accL[cf], false, false);
    }
    wait_async0();
    __syncthreads();
  }

  // --- epilogue: bias + SiLU(gate)*lin -> bf16 act rows
  #pragma unroll
  for (int cf = 0; cf < 2; ++cf) {
    int ncol = c0 + wn * 32 + cf * 16 + ln;
    float bg = b1[(size_t)e * 2 * F_ + ncol];
    float bl = b1[(size_t)e * 2 * F_ + F_ + ncol];
    #pragma unroll
    for (int j = 0; j < 8; ++j) {
      int mrow = wm * 16 + j + half * 8;
      int p = sPair[mrow];
      if (p < 0) continue;
      float g = accG[cf][j] + bg;
      float l = accL[cf][j] + bl;
      float s = (g / (1.0f + __expf(-g))) * l;
      act[(size_t)p * F_ + ncol] = f2bf(s);
    }
  }
}

// ---------------------------------------------------------------------------
// K3: GEMM2: out[t,h] += scale_p * (act[p]@w2[e]^T + b2[e])  via f32 atomics
// ---------------------------------------------------------------------------
__global__ __launch_bounds__(256) void moe_gemm2(
    const unsigned short* __restrict__ act,   // [T*K][F] bf16
    const float* __restrict__ w2,             // [E][H][F] f32
    const float* __restrict__ b2,             // [E][H] f32
    const float* __restrict__ scal,           // [T*K] f32
    const int* __restrict__ cnt,
    const int* __restrict__ list,
    float* __restrict__ out)                  // [T][H] f32
{
  const int e   = blockIdx.z;
  const int n_e = cnt[e];
  const int rT  = blockIdx.y;
  if (rT * BM >= n_e) return;
  const int h0  = blockIdx.x * 64;

  __shared__ unsigned short sA[2][BM * PA];
  __shared__ unsigned short sB[2][64 * PB];
  __shared__ int sPair[BM];

  const int tid = threadIdx.x;
  if (tid < BM) {
    int r = rT * BM + tid;
    sPair[tid] = (r < n_e) ? list[e * CAP + r] : -1;
  }
  __syncthreads();

  const int aRow = tid >> 2, aSeg = tid & 3;
  int pA = sPair[aRow];
  const unsigned short* aSrc =
      act + (size_t)(pA < 0 ? 0 : pA) * F_ + aSeg * 8;

  const int bRow = tid >> 2, bCh = tid & 3;        // B: 8 f32 per thread
  const float* bSrc =
      w2 + (size_t)e * H_ * F_ + (size_t)(h0 + bRow) * F_ + bCh * 8;

  const int wave = tid >> 5, lane = tid & 31;
  const int wm = wave >> 1, wn = wave & 1;
  const int half = lane >> 4, ln = lane & 15;

  floatx8 acc[2] = {};

  auto stage = [&](int buf, int k0) {
    async_cp16(&sA[buf][aRow * PA + aSeg * 8], aSrc + k0);
    __builtin_prefetch(bSrc + k0 + 2 * BK, 0, 0);
    const F4* s4 = (const F4*)(bSrc + k0);
    alignas(16) unsigned short tmp[8];
    #pragma unroll
    for (int q = 0; q < 2; ++q) {
      F4 f = s4[q];
      tmp[4 * q + 0] = f2bf(f.x); tmp[4 * q + 1] = f2bf(f.y);
      tmp[4 * q + 2] = f2bf(f.z); tmp[4 * q + 3] = f2bf(f.w);
    }
    *(U4*)(&sB[buf][bRow * PB + bCh * 8]) = *(const U4*)(&tmp[0]);
  };

  stage(0, 0);
  wait_async0();
  __syncthreads();

  const int nk = F_ / BK;
  for (int i = 0; i < nk; ++i) {
    const int buf = i & 1;
    if (i + 1 < nk) stage(buf ^ 1, (i + 1) * BK);

    Frag fa;
    {
      int m = wm * 16 + ln;
      fa.q[0] = *(const U4*)(&sA[buf][m * PA + half * 16]);
      fa.q[1] = *(const U4*)(&sA[buf][m * PA + half * 16 + 8]);
    }
    #pragma unroll
    for (int cf = 0; cf < 2; ++cf) {
      int nb = wn * 32 + cf * 16 + ln;
      Frag fb;
      fb.q[0] = *(const U4*)(&sB[buf][nb * PB + half * 16 + 0]);
      fb.q[1] = *(const U4*)(&sB[buf][nb * PB + half * 16 + 8]);
      acc[cf] = __builtin_amdgcn_wmma_f32_16x16x32_bf16(
          false, fa.v, false, fb.v, (short)0, acc[cf], false, false);
    }
    wait_async0();
    __syncthreads();
  }

  // --- epilogue: bias, scale, atomic accumulate into out
  #pragma unroll
  for (int cf = 0; cf < 2; ++cf) {
    int h = h0 + wn * 32 + cf * 16 + ln;
    float bb = b2[(size_t)e * H_ + h];
    #pragma unroll
    for (int j = 0; j < 8; ++j) {
      int mrow = wm * 16 + j + half * 8;
      int p = sPair[mrow];
      if (p < 0) continue;
      int t = p >> 1;
      float val = scal[p] * (acc[cf][j] + bb);
      atomicAdd(&out[(size_t)t * H_ + h], val);
    }
  }
}

// ---------------------------------------------------------------------------
// Launcher.  ws layout: [cnt 256B][list 64KB][X16 2MB][act 8MB] ~= 10.1 MB
// ---------------------------------------------------------------------------
extern "C" void kernel_launch(void* const* d_in, const int* in_sizes, int n_in,
                              void* d_out, int out_size, void* d_ws, size_t ws_size,
                              hipStream_t stream) {
  const float* x    = (const float*)d_in[0];   // hidden_states (T,H)
  const int*   sel  = (const int*)  d_in[1];   // token_selected_experts (T,K)
  const float* scal = (const float*)d_in[2];   // token_final_scales (T,K)
  const float* w1   = (const float*)d_in[3];   // (E, 2F, H)
  const float* b1   = (const float*)d_in[4];   // (E, 2F)
  const float* w2   = (const float*)d_in[5];   // (E, H, F)
  const float* b2   = (const float*)d_in[6];   // (E, H)
  float* out = (float*)d_out;

  char* ws = (char*)d_ws;
  int*            cnt  = (int*)ws;                                   // 256 B
  int*            list = (int*)(ws + 256);                           // E*CAP*4 = 64 KB
  unsigned short* X16  = (unsigned short*)(ws + 256 + 65536);        // T*H*2 = 2 MB
  unsigned short* act  = (unsigned short*)(ws + 256 + 65536 + (size_t)T_ * H_ * 2);

  moe_init <<<2048, 256, 0, stream>>>(x, out, X16, cnt);
  moe_route<<<(T_ * KSEL + 255) / 256, 256, 0, stream>>>(sel, cnt, list);
  moe_gemm1<<<dim3(F_ / 64, CAP / BM, E_), 256, 0, stream>>>(X16, w1, b1, cnt, list, act);
  moe_gemm2<<<dim3(H_ / 64, CAP / BM, E_), 256, 0, stream>>>(act, w2, b2, scal, cnt, list, out);
}
